// GraphTransformerLayer_27522150433320
// MI455X (gfx1250) — matrix operand order repaired and checked by reading, not compiled, not run on hardware
//
#include <hip/hip_runtime.h>
#include <hip/hip_bf16.h>
#include <math.h>

// ---------------------------------------------------------------------------
// Graph transformer layer for MI455X (gfx1250), wave32 + WMMA f16->f32,
// async global->LDS staging (ASYNCcnt) for GEMM B-tiles and attention V-tiles.
// ---------------------------------------------------------------------------

typedef __attribute__((ext_vector_type(16))) _Float16 v16h;
typedef __attribute__((ext_vector_type(8)))  float    v8f;

#define NTOK 4096
#define TDIM 512
#define NHEADS 8
#define HD 64
#define TFFN 2048
#define QKVD 1536

// ---- gfx1250 async global->LDS path (guarded; sync fallback) ---------------
#if defined(__AMDGCN__) && __has_builtin(__builtin_amdgcn_global_load_async_to_lds_b128)
#define HAVE_ASYNC 1
#else
#define HAVE_ASYNC 0
#endif

#if HAVE_ASYNC
typedef int v4i_gcc __attribute__((__vector_size__(16)));
typedef __attribute__((address_space(1))) v4i_gcc gas_v4i;
typedef __attribute__((address_space(3))) v4i_gcc las_v4i;
__device__ __forceinline__ void async_ldg_b128(const _Float16* g, _Float16* l) {
  __builtin_amdgcn_global_load_async_to_lds_b128(
      (gas_v4i*)g, (las_v4i*)l, 0, 0);
}
__device__ __forceinline__ void wait_async0() {
#if __has_builtin(__builtin_amdgcn_s_wait_asynccnt)
  __builtin_amdgcn_s_wait_asynccnt(0);
#else
  asm volatile("s_wait_asynccnt 0x0" ::: "memory");
#endif
}
__device__ __forceinline__ void wait_async1() {
#if __has_builtin(__builtin_amdgcn_s_wait_asynccnt)
  __builtin_amdgcn_s_wait_asynccnt(1);
#else
  asm volatile("s_wait_asynccnt 0x1" ::: "memory");
#endif
}
#else
__device__ __forceinline__ void wait_async0() {}
__device__ __forceinline__ void wait_async1() {}
#endif

// --- WMMA fragment helpers (contiguous-k order, wave32) --------------
// A (16x32 f16): lane L -> row m=L%16; half=L/16.
//   elem i : k = half*16 + i
__device__ __forceinline__ v16h load_a_frag(const _Float16* row, int half) {
  v16h a;
  const _Float16* p0 = row + half * 16;
#pragma unroll
  for (int i = 0; i < 16; ++i) { a[i] = p0[i]; }
  return a;
}

// B (32x16 f16): lane L -> col n=L%16; elem i -> k = (L/16)*16 + i
__device__ __forceinline__ v16h load_b_frag(const _Float16* p) {
  v16h b;
#pragma unroll
  for (int i = 0; i < 16; ++i) b[i] = p[i];
  return b;
}

__device__ __forceinline__ v16h load_b_frag_strided(const _Float16* p, int stride) {
  v16h b;
#pragma unroll
  for (int i = 0; i < 16; ++i) b[i] = p[i * stride];
  return b;
}

__device__ __forceinline__ v8f wmma_f16(v16h a, v16h b, v8f c) {
  return __builtin_amdgcn_wmma_f32_16x16x32_f16(false, a, false, b,
                                                (short)0, c, false, false);
}

// reduce across the 16 lanes of a half-wave (rows of C layout)
__device__ __forceinline__ float rowmax16(float v) {
#pragma unroll
  for (int m = 1; m <= 8; m <<= 1) v = fmaxf(v, __shfl_xor(v, m, 32));
  return v;
}
__device__ __forceinline__ float rowsum16(float v) {
#pragma unroll
  for (int m = 1; m <= 8; m <<= 1) v += __shfl_xor(v, m, 32);
  return v;
}

// --- weight conversion ------------------------------------------------------
__global__ void cvt_f32_to_f16(const float* __restrict__ in,
                               _Float16* __restrict__ out, int n) {
  int i = blockIdx.x * 256 + threadIdx.x;
  if (i < n) out[i] = (_Float16)in[i];
}

// --- layernorm: one row (512) per 256-thread block --------------------------
__global__ __launch_bounds__(256) void layernorm_f16(
    const float* __restrict__ x, const float* __restrict__ g,
    const float* __restrict__ b, _Float16* __restrict__ out) {
  __shared__ float red[256];
  const int row = blockIdx.x, t = threadIdx.x;
  const float v0 = x[(size_t)row * TDIM + t];
  const float v1 = x[(size_t)row * TDIM + 256 + t];
  red[t] = v0 + v1;
  __syncthreads();
#pragma unroll
  for (int o = 128; o > 0; o >>= 1) {
    if (t < o) red[t] += red[t + o];
    __syncthreads();
  }
  const float mean = red[0] * (1.0f / TDIM);
  __syncthreads();
  const float d0 = v0 - mean, d1 = v1 - mean;
  red[t] = d0 * d0 + d1 * d1;
  __syncthreads();
#pragma unroll
  for (int o = 128; o > 0; o >>= 1) {
    if (t < o) red[t] += red[t + o];
    __syncthreads();
  }
  const float rstd = rsqrtf(red[0] * (1.0f / TDIM) + 1e-5f);
  out[(size_t)row * TDIM + t]       = (_Float16)(d0 * rstd * g[t] + b[t]);
  out[(size_t)row * TDIM + 256 + t] = (_Float16)(d1 * rstd * g[256 + t] + b[256 + t]);
}

// --- generic GEMM: C[M,N] = A[M,K] @ B[N,K]^T + bias (+gelu) (+resid) -------
// wave tile 32x64 (8 WMMA / 32-K step), 8 waves/block -> block tile 256x64.
// B tile (64 cols x 32 k, shared by all 8 waves) is double-buffered in LDS,
// filled with async global->LDS DMA and consumed with plain DS loads.
template <int ACT>
__global__ __launch_bounds__(256) void gemm_wmma(
    const _Float16* __restrict__ A, const _Float16* __restrict__ B,
    const float* __restrict__ bias, const float* __restrict__ resid,
    float* __restrict__ outf, _Float16* __restrict__ outh,
    int M, int N, int K) {
  __shared__ __align__(16) _Float16 sB[2][64 * 32];

  const int tid  = threadIdx.x;
  const int lane = tid & 31;
  const int w    = tid >> 5;
  const int half = lane >> 4;
  const int mn   = lane & 15;
  const int r0 = blockIdx.x * 256 + w * 32;
  const int c0 = blockIdx.y * 64;

  v8f acc[2][4] = {};
  const _Float16* arow0 = A + (size_t)(r0 + mn) * K;
  const _Float16* arow1 = A + (size_t)(r0 + 16 + mn) * K;

  // B staging: thread t moves 16B: col bn = t/4, k-halves bk = (t%4)*8
  const int bn = tid >> 2;
  const int bk = (tid & 3) * 8;
  const _Float16* bsrc = B + (size_t)(c0 + bn) * K + bk;
  const int ldsoff = bn * 32 + bk;

  const int ntiles = K >> 5;

  // prefetch tile 0
  {
#if HAVE_ASYNC
    async_ldg_b128(bsrc, &sB[0][ldsoff]);
#else
    *(uint4*)&sB[0][ldsoff] = *(const uint4*)bsrc;
#endif
  }

  for (int it = 0; it < ntiles; ++it) {
    const int cur = it & 1;
    const int kk = it << 5;
    const bool more = (it + 1 < ntiles);
    if (more) {
#if HAVE_ASYNC
      async_ldg_b128(bsrc + kk + 32, &sB[cur ^ 1][ldsoff]);
#else
      *(uint4*)&sB[cur ^ 1][ldsoff] = *(const uint4*)(bsrc + kk + 32);
#endif
      wait_async1();   // tile `it` (our contribution) complete
    } else {
      wait_async0();
    }
    __syncthreads();   // whole tile `it` visible to all waves

    const v16h a0 = load_a_frag(arow0 + kk, half);
    const v16h a1 = load_a_frag(arow1 + kk, half);
#pragma unroll
    for (int t = 0; t < 4; ++t) {
      const v16h b = load_b_frag(&sB[cur][(t * 16 + mn) * 32 + half * 16]);
      acc[0][t] = wmma_f16(a0, b, acc[0][t]);
      acc[1][t] = wmma_f16(a1, b, acc[1][t]);
    }
    __syncthreads();   // everyone done reading `cur` before it is overwritten
  }

#pragma unroll
  for (int rt = 0; rt < 2; ++rt)
#pragma unroll
    for (int t = 0; t < 4; ++t) {
      const int col = c0 + t * 16 + mn;
      const float bv = bias[col];
#pragma unroll
      for (int j = 0; j < 8; ++j) {
        const int row = r0 + rt * 16 + half * 8 + j;
        float v = acc[rt][t][j] + bv;
        if (ACT == 1) v = 0.5f * v * (1.0f + erff(v * 0.70710678118654752f));
        if (resid) v += resid[(size_t)row * N + col];
        if (outf) outf[(size_t)row * N + col] = v;
        if (outh) outh[(size_t)row * N + col] = (_Float16)v;
      }
    }
}

// --- flash attention with block-diagonal (batch_index) mask -----------------
// grid = (HEADS, NTOK/64); block = 128 (4 waves); wave: one 16-row q tile.
// V tile staged per-wave via async global->LDS, overlapped with score WMMAs.
__global__ __launch_bounds__(128) void flash_attn(
    const _Float16* __restrict__ qkv, const int* __restrict__ bidx,
    _Float16* __restrict__ ctx) {
  __shared__ int s_bi[NTOK];
  __shared__ __align__(16) _Float16 s_v[4][32 * 64];
  __shared__ __align__(16) _Float16 s_p[4][16 * 32];

  const int tid = threadIdx.x, lane = tid & 31, w = tid >> 5;
  const int head = blockIdx.x;
  const int r0 = (blockIdx.y * 4 + w) * 16;
  const int half = lane >> 4, mn = lane & 15;

  for (int i = tid; i < NTOK; i += 128) s_bi[i] = bidx[i];
  __syncthreads();

  const _Float16* qrow = qkv + (size_t)(r0 + mn) * QKVD + head * HD;
  const v16h aq0 = load_a_frag(qrow, half);        // d = 0..31
  const v16h aq1 = load_a_frag(qrow + 32, half);   // d = 32..63

  int bq[8];
#pragma unroll
  for (int j = 0; j < 8; ++j) bq[j] = s_bi[r0 + half * 8 + j];

  float mrow[8], lrow[8];
#pragma unroll
  for (int j = 0; j < 8; ++j) { mrow[j] = -1e30f; lrow[j] = 0.0f; }
  v8f o[4] = {};
  const float scale = 0.125f;  // 1/sqrt(64)

  for (int kb = 0; kb < NTOK; kb += 32) {
    // stage V tile (32 keys x 64 dims = 4KB) into per-wave LDS: lane owns row
    {
      const _Float16* vsrc = qkv + (size_t)(kb + lane) * QKVD + 2 * TDIM + head * HD;
      _Float16* vdst = &s_v[w][lane * 64];
#if HAVE_ASYNC
#pragma unroll
      for (int c = 0; c < 8; ++c) async_ldg_b128(vsrc + c * 8, vdst + c * 8);
#else
#pragma unroll
      for (int c = 0; c < 8; ++c) ((uint4*)vdst)[c] = ((const uint4*)vsrc)[c];
#endif
    }

    // scores S = q @ k^T : two 16x16 tiles (keys kb..+15, kb+16..+31)
    v8f s0 = {}, s1 = {};
    const _Float16* kb0 = qkv + (size_t)(kb + mn) * QKVD + TDIM + head * HD;
    const _Float16* kb1 = kb0 + (size_t)16 * QKVD;
    s0 = wmma_f16(aq0, load_b_frag(kb0 + half * 16), s0);
    s0 = wmma_f16(aq1, load_b_frag(kb0 + 32 + half * 16), s0);
    s1 = wmma_f16(aq0, load_b_frag(kb1 + half * 16), s1);
    s1 = wmma_f16(aq1, load_b_frag(kb1 + 32 + half * 16), s1);

    // mask + online softmax (row stats shared across the 16 lanes of a half)
    const int bk0 = s_bi[kb + mn];
    const int bk1 = s_bi[kb + 16 + mn];
    float p0[8], p1[8], corr[8];
#pragma unroll
    for (int j = 0; j < 8; ++j) {
      const bool a0 = (bq[j] == bk0), a1 = (bq[j] == bk1);
      const float v0 = a0 ? s0[j] * scale : -1e30f;
      const float v1 = a1 ? s1[j] * scale : -1e30f;
      const float tmax = rowmax16(fmaxf(v0, v1));
      const float mnew = fmaxf(mrow[j], tmax);
      const float c = __expf(mrow[j] - mnew);
      const float e0 = a0 ? __expf(v0 - mnew) : 0.0f;
      const float e1 = a1 ? __expf(v1 - mnew) : 0.0f;
      lrow[j] = lrow[j] * c + rowsum16(e0 + e1);
      mrow[j] = mnew;
      corr[j] = c;
      p0[j] = e0; p1[j] = e1;
    }
#pragma unroll
    for (int t = 0; t < 4; ++t)
#pragma unroll
      for (int j = 0; j < 8; ++j) o[t][j] *= corr[j];

    // re-layout P (C layout -> A layout) through per-wave LDS
#pragma unroll
    for (int j = 0; j < 8; ++j) {
      const int m = half * 8 + j;
      s_p[w][m * 32 + mn]      = (_Float16)p0[j];
      s_p[w][m * 32 + 16 + mn] = (_Float16)p1[j];
    }
    __builtin_amdgcn_wave_barrier();
    asm volatile("" ::: "memory");  // same-wave LDS ops are in-order (DScnt)

    wait_async0();                  // V tile resident before P@V
    const v16h ap = load_a_frag(&s_p[w][mn * 32], half);
#pragma unroll
    for (int dc = 0; dc < 4; ++dc) {
      const v16h bv =
          load_b_frag_strided(&s_v[w][(half * 16) * 64 + dc * 16 + mn], 64);
      o[dc] = wmma_f16(ap, bv, o[dc]);
    }
    __builtin_amdgcn_wave_barrier();
  }

  // normalize and write ctx (f16) in [n, DIM] layout
#pragma unroll
  for (int dc = 0; dc < 4; ++dc)
#pragma unroll
    for (int j = 0; j < 8; ++j) {
      const int m = half * 8 + j;
      const float v = o[dc][j] / lrow[j];
      ctx[(size_t)(r0 + m) * TDIM + head * HD + dc * 16 + mn] = (_Float16)v;
    }
}

// ---------------------------------------------------------------------------
extern "C" void kernel_launch(void* const* d_in, const int* in_sizes, int n_in,
                              void* d_out, int out_size, void* d_ws, size_t ws_size,
                              hipStream_t stream) {
  const float* x      = (const float*)d_in[0];
  const int*   bidx   = (const int*)d_in[1];
  const float* ln1_g  = (const float*)d_in[2];
  const float* ln1_b  = (const float*)d_in[3];
  const float* w_qkv  = (const float*)d_in[4];
  const float* b_qkv  = (const float*)d_in[5];
  const float* w_out  = (const float*)d_in[6];
  const float* b_out  = (const float*)d_in[7];
  const float* ln2_g  = (const float*)d_in[8];
  const float* ln2_b  = (const float*)d_in[9];
  const float* w1     = (const float*)d_in[10];
  const float* b1     = (const float*)d_in[11];
  const float* w2     = (const float*)d_in[12];
  const float* b2     = (const float*)d_in[13];
  float* out = (float*)d_out;

  char* ws = (char*)d_ws;  // needs ~56 MiB
  _Float16* h16   = (_Float16*)(ws);
  _Float16* qkv16 = (_Float16*)(ws + (size_t)(4  << 20));
  _Float16* ctx16 = (_Float16*)(ws + (size_t)(16 << 20));
  float*    x1    = (float*)   (ws + (size_t)(20 << 20));
  _Float16* h2    = (_Float16*)(ws + (size_t)(28 << 20));
  _Float16* act16 = (_Float16*)(ws + (size_t)(32 << 20));
  _Float16* wq16  = (_Float16*)(ws + (size_t)(48 << 20));
  _Float16* wo16  = (_Float16*)(ws + (size_t)(50 << 20));
  _Float16* w116  = (_Float16*)(ws + (size_t)(51 << 20));
  _Float16* w216  = (_Float16*)(ws + (size_t)(53 << 20));

  // weights -> f16
  cvt_f32_to_f16<<<(QKVD * TDIM) / 256, 256, 0, stream>>>(w_qkv, wq16, QKVD * TDIM);
  cvt_f32_to_f16<<<(TDIM * TDIM) / 256, 256, 0, stream>>>(w_out, wo16, TDIM * TDIM);
  cvt_f32_to_f16<<<(TFFN * TDIM) / 256, 256, 0, stream>>>(w1, w116, TFFN * TDIM);
  cvt_f32_to_f16<<<(TDIM * TFFN) / 256, 256, 0, stream>>>(w2, w216, TDIM * TFFN);

  // ln1
  layernorm_f16<<<NTOK, 256, 0, stream>>>(x, ln1_g, ln1_b, h16);

  // qkv = h @ Wqkv^T + b
  gemm_wmma<0><<<dim3(NTOK / 256, QKVD / 64), 256, 0, stream>>>(
      h16, wq16, b_qkv, nullptr, nullptr, qkv16, NTOK, QKVD, TDIM);

  // flash attention -> ctx16
  flash_attn<<<dim3(NHEADS, NTOK / 64), 128, 0, stream>>>(qkv16, bidx, ctx16);

  // x1 = ctx @ Wout^T + b_out + x
  gemm_wmma<0><<<dim3(NTOK / 256, TDIM / 64), 256, 0, stream>>>(
      ctx16, wo16, b_out, x, x1, nullptr, NTOK, TDIM, TDIM);

  // ln2
  layernorm_f16<<<NTOK, 256, 0, stream>>>(x1, ln2_g, ln2_b, h2);

  // act = gelu(h2 @ W1^T + b1)
  gemm_wmma<1><<<dim3(NTOK / 256, TFFN / 64), 256, 0, stream>>>(
      h2, w116, b1, nullptr, nullptr, act16, NTOK, TFFN, TDIM);

  // out = act @ W2^T + b2 + x1
  gemm_wmma<0><<<dim3(NTOK / 256, TDIM / 64), 256, 0, stream>>>(
      act16, w216, b2, x1, out, nullptr, NTOK, TDIM, TFFN);
}
